// QRNN_80290118631664
// MI455X (gfx1250) — matrix operand order repaired and checked by reading, not compiled, not run on hardware
//
#include <hip/hip_runtime.h>
#include <hip/hip_bf16.h>
#include <math.h>

#define Tdim 1024
#define Hdim 1024
#define N3H  3072
#define Bdim 16
#define NLay 3
#define AST  17   // sA row stride in dwords (34 f16 = 32 data + 1 dword pad)
#define WST  17   // sW c-stride in dwords  (34 f16 = 32 data + 1 dword pad)

typedef _Float16 f16;
typedef __attribute__((ext_vector_type(16))) _Float16 v16h;
typedef __attribute__((ext_vector_type(2)))  float    v2f;
typedef __attribute__((ext_vector_type(8)))  float    v8f;

union Frag16 { v16h h; float f[8]; };

#if defined(__has_builtin)
# if __has_builtin(__builtin_amdgcn_tensor_load_to_lds)
#  define HAVE_TDM 1
# endif
#endif
#ifndef HAVE_TDM
# define HAVE_TDM 0
#endif

#if HAVE_TDM
# define NBUF 2
typedef unsigned int __attribute__((ext_vector_type(4))) v4u;
typedef int          __attribute__((ext_vector_type(8))) v8i;
typedef int          __attribute__((ext_vector_type(4))) v4i;

// Build a Tensor DMA Descriptor (D#, ISA 8.3-8.6) and issue TENSOR_LOAD_TO_LDS.
// data_size = 2B; LDS padding: +1 dword after every 16 dwords (-> 17-dword rows).
__device__ __forceinline__ void tdm_load_tile(unsigned int lds_addr, const void* gptr,
                                              unsigned int tile0, unsigned int tile1,
                                              unsigned int tile2,
                                              unsigned long long stride0,  // Y line stride (elems)
                                              unsigned long long stride1)  // Z stride (elems)
{
    const unsigned long long ga = (unsigned long long)(uintptr_t)gptr;
    v4u g0;
    g0.x = 1u;                                            // count=1, user D#
    g0.y = lds_addr;                                      // lds_addr [63:32]
    g0.z = (unsigned int)ga;                              // global_addr [95:64]
    g0.w = ((unsigned int)(ga >> 32) & 0x01FFFFFFu) | 0x80000000u;  // [120:96]+type=2

    unsigned int w[8] = {0, 0, 0, 0, 0, 0, 0, 0};
    w[0] = (1u << 16)     // data_size = 2 bytes
         | (1u << 20)     // pad_enable
         | (3u << 22);    // pad_interval = 16 dwords (pad_amount=0 -> 1 dword)
    w[1] |= (tile0 & 0xFFFFu) << 16;                      // tensor_dim0 [79:48]
    w[2] |= (tile0 >> 16) & 0xFFFFu;
    w[2] |= (tile1 & 0xFFFFu) << 16;                      // tensor_dim1 [111:80]
    w[3] |= (tile1 >> 16) & 0xFFFFu;
    w[3] |= (tile0 & 0xFFFFu) << 16;                      // tile_dim0 [127:112]
    w[4] |= (tile1 & 0xFFFFu);                            // tile_dim1 [143:128]
    w[4] |= (tile2 & 0xFFFFu) << 16;                      // tile_dim2 [159:144]
    w[5]  = (unsigned int)(stride0 & 0xFFFFFFFFull);      // dim0_stride [207:160]
    w[6] |= (unsigned int)((stride0 >> 32) & 0xFFFFull);
    w[6] |= (unsigned int)(stride1 & 0xFFFFull) << 16;    // dim1_stride [255:208]
    w[7]  = (unsigned int)((stride1 >> 16) & 0xFFFFFFFFull);
    v8i g1 = { (int)w[0], (int)w[1], (int)w[2], (int)w[3],
               (int)w[4], (int)w[5], (int)w[6], (int)w[7] };
    v4i g2 = { (int)tile2, 0, 0, 0 };                     // tensor_dim2 (OOB only)
    v4i g3 = { 0, 0, 0, 0 };
    v8i g4 = { 0, 0, 0, 0, 0, 0, 0, 0 };
    __builtin_amdgcn_tensor_load_to_lds(g0, g1, g2, g3, g4, 0);
}
#else
# define NBUF 1
#endif

// ---------------------------------------------------------------------------
__global__ __launch_bounds__(256) void embed_kernel(const int* __restrict__ ids,
                                                    const float* __restrict__ emb,
                                                    float* __restrict__ x) {
    const int r = blockIdx.x;                 // 0 .. B*T-1
    const int id = ids[r];
    const float4* src = (const float4*)(emb + (size_t)id * Hdim);
    float4* dst = (float4*)(x + (size_t)r * Hdim);
    dst[threadIdx.x] = src[threadIdx.x];
}

__global__ __launch_bounds__(256) void zero_kernel(float* p, int n) {
    int g = blockIdx.x * 256 + threadIdx.x;
    if (g < n) p[g] = 0.0f;
}

// ---------------------------------------------------------------------------
// x: fp32 -> f16 hi + residual lo (same layout)
// ---------------------------------------------------------------------------
__global__ __launch_bounds__(256) void split_f16_kernel(const float* __restrict__ src,
                                                        f16* __restrict__ hi,
                                                        f16* __restrict__ lo, int n) {
    const int g = blockIdx.x * 256 + threadIdx.x;
    if (g < n) {
        const float v = src[g];
        const f16 h = (f16)v;
        hi[g] = h;
        lo[g] = (f16)(v - (float)h);
    }
}

// ---------------------------------------------------------------------------
// convW (ik, c, n) fp32 -> transposed f16 hi/lo (ik, n, c): TDM-friendly tiles
// with contiguous inner c. 32x32 LDS tile transpose, coalesced both sides.
// ---------------------------------------------------------------------------
__global__ __launch_bounds__(256) void wsplit_transpose_kernel(const float* __restrict__ W,
                                                               f16* __restrict__ Wth,
                                                               f16* __restrict__ Wtl) {
    const int nb = blockIdx.x;                 // 3H/32
    const int cb = blockIdx.y;                 // H/32
    const int ik = blockIdx.z;                 // NLay*K
    __shared__ float tile[32][33];
    const int tn = threadIdx.x & 31;
    const int tr = threadIdx.x >> 5;           // 8 rows per pass
    const size_t ibase = (size_t)ik * Hdim * N3H;
    #pragma unroll
    for (int cc = tr; cc < 32; cc += 8)
        tile[cc][tn] = W[ibase + (size_t)(cb * 32 + cc) * N3H + nb * 32 + tn];
    __syncthreads();
    const size_t obase = (size_t)ik * N3H * Hdim;
    #pragma unroll
    for (int nn = tr; nn < 32; nn += 8) {
        const float v = tile[tn][nn];          // [c_local][n_local]
        const f16 h = (f16)v;
        const size_t o = obase + (size_t)(nb * 32 + nn) * Hdim + cb * 32 + tn;
        Wth[o] = h;
        Wtl[o] = (f16)(v - (float)h);
    }
}

// ---------------------------------------------------------------------------
// Causal conv1d (K=3) as split-precision f16 WMMA GEMM (f32 accumulate),
// tiles staged by the Tensor Data Mover, double-buffered against WMMA.
// Block: 8 waves; tile 64(M) x 128(N); 32-channel K chunks.
// ---------------------------------------------------------------------------
__global__ __launch_bounds__(256) void conv_wmma_f16_kernel(const f16* __restrict__ xh,
                                                            const f16* __restrict__ xl,
                                                            const f16* __restrict__ Wth,
                                                            const f16* __restrict__ Wtl,
                                                            float* __restrict__ co) {
    const int bm = blockIdx.x;                 // 64-row M tile (one batch seq)
    const int bn = blockIdx.y;                 // 128-col N tile
    const int b  = (bm * 64) / Tdim;
    const int t0 = (bm * 64) % Tdim;
    const int n0 = bn * 128;

    __shared__ float sAh[NBUF][66 * AST];
    __shared__ float sAl[NBUF][66 * AST];
    __shared__ float sWh[NBUF][3 * 128 * WST];   // [k][n][c-pair]
    __shared__ float sWl[NBUF][3 * 128 * WST];

    const int tid  = threadIdx.x;
    const int lane = tid & 31;
    const int wave = tid >> 5;
    const int wm   = wave & 3;
    const int wn   = wave >> 2;
    const int hl   = lane >> 4;
    const int mloc = wm * 16 + (lane & 15);

    if (t0 == 0 && tid < 34) {                 // zero causal halo rows 0..1
        #pragma unroll
        for (int nb2 = 0; nb2 < NBUF; ++nb2) {
            sAh[nb2][tid] = 0.0f;
            sAl[nb2][tid] = 0.0f;
        }
    }

    v8f acc[4] = {};

#if HAVE_TDM
    auto issue = [&](int c0, int buf) {
        const int  arow0 = (t0 == 0) ? 0 : (t0 - 2);
        const unsigned nrows = (t0 == 0) ? 64u : 66u;
        const unsigned ldsSk = (t0 == 0) ? (2u * AST * 4u) : 0u;
        tdm_load_tile((unsigned)(uintptr_t)&sAh[buf][0] + ldsSk,
                      xh + (size_t)(b * Tdim + arow0) * Hdim + c0,
                      32u, nrows, 0u, Hdim, 0ull);
        tdm_load_tile((unsigned)(uintptr_t)&sAl[buf][0] + ldsSk,
                      xl + (size_t)(b * Tdim + arow0) * Hdim + c0,
                      32u, nrows, 0u, Hdim, 0ull);
        tdm_load_tile((unsigned)(uintptr_t)&sWh[buf][0],
                      Wth + (size_t)n0 * Hdim + c0,
                      32u, 128u, 3u, Hdim, (unsigned long long)N3H * Hdim);
        tdm_load_tile((unsigned)(uintptr_t)&sWl[buf][0],
                      Wtl + (size_t)n0 * Hdim + c0,
                      32u, 128u, 3u, Hdim, (unsigned long long)N3H * Hdim);
    };
    if (wave == 0) issue(0, 0);
#endif

    for (int ci = 0; ci < Hdim / 32; ++ci) {
        const int buf = (NBUF == 2) ? (ci & 1) : 0;
#if HAVE_TDM
        if (wave == 0) {
            if (ci + 1 < Hdim / 32) {
                issue((ci + 1) * 32, buf ^ 1);             // overlap next chunk DMA
                __builtin_amdgcn_s_wait_tensorcnt(4);      // oldest 4 (this chunk) done
            } else {
                __builtin_amdgcn_s_wait_tensorcnt(0);
            }
        }
#else
        const int c0 = ci * 32;
        for (int idx = tid; idx < 66 * 16; idx += 256) {
            const int tt = idx >> 4, cp = idx & 15;
            const int gt = t0 - 2 + tt;
            float vh = 0.0f, vl = 0.0f;
            if (gt >= 0) {
                const size_t off = ((size_t)(b * Tdim + gt) * Hdim + c0) >> 1;
                vh = ((const float*)xh)[off + cp];
                vl = ((const float*)xl)[off + cp];
            }
            sAh[buf][tt * AST + cp] = vh;
            sAl[buf][tt * AST + cp] = vl;
        }
        for (int idx = tid; idx < 3 * 128 * 16; idx += 256) {
            const int cp = idx & 15;
            const int nn = (idx >> 4) & 127;
            const int kk = idx >> 11;
            const size_t off = ((size_t)kk * N3H * Hdim + (size_t)(n0 + nn) * Hdim + c0) >> 1;
            sWh[buf][(kk * 128 + nn) * WST + cp] = ((const float*)Wth)[off + cp];
            sWl[buf][(kk * 128 + nn) * WST + cp] = ((const float*)Wtl)[off + cp];
        }
#endif
        __syncthreads();

        #pragma unroll
        for (int k = 0; k < 3; ++k) {
            Frag16 ah, al;                      // A 16x32: dword v -> K={2v+8h,+1}(+8,v>=4)
            const int arow = (mloc + k) * AST;
            #pragma unroll
            for (int v = 0; v < 8; ++v) {
                const int dc = v + 4 * hl + (v >= 4 ? 4 : 0);
                ah.f[v] = sAh[buf][arow + dc];
                al.f[v] = sAl[buf][arow + dc];
            }
            #pragma unroll
            for (int nt = 0; nt < 4; ++nt) {
                const int nn = wn * 64 + nt * 16 + (lane & 15);
                Frag16 bh, bl;                  // B 32x16: dword v -> K={2v+8h,+1}(+8,v>=4)
                const int bbase = (k * 128 + nn) * WST;
                #pragma unroll
                for (int v = 0; v < 8; ++v) {
                    bh.f[v] = sWh[buf][bbase + v + 4 * hl + (v >= 4 ? 4 : 0)];
                    bl.f[v] = sWl[buf][bbase + v + 4 * hl + (v >= 4 ? 4 : 0)];
                }
                acc[nt] = __builtin_amdgcn_wmma_f32_16x16x32_f16(
                    false, ah.h, false, bh.h, (short)0, acc[nt], false, false);
                acc[nt] = __builtin_amdgcn_wmma_f32_16x16x32_f16(
                    false, ah.h, false, bl.h, (short)0, acc[nt], false, false);
                acc[nt] = __builtin_amdgcn_wmma_f32_16x16x32_f16(
                    false, al.h, false, bh.h, (short)0, acc[nt], false, false);
            }
        }
        __syncthreads();
    }

    const int gm = bm * 64 + wm * 16;
    #pragma unroll
    for (int nt = 0; nt < 4; ++nt) {
        const int ncol = n0 + wn * 64 + nt * 16 + (lane & 15);
        #pragma unroll
        for (int v = 0; v < 8; ++v) {
            const int row = v + 8 * hl;         // C/D: M = v + 8*(lane/16)
            co[(size_t)(gm + row) * N3H + ncol] = acc[nt][v];
        }
    }
}

// ---------------------------------------------------------------------------
__global__ __launch_bounds__(256) void qrnn_scan_kernel(const float* __restrict__ co,
                                                        const float* __restrict__ bias,
                                                        float* __restrict__ h0,
                                                        float* __restrict__ hbuf) {
    const int g = blockIdx.x * 256 + threadIdx.x;   // 0 .. B*H-1
    const int b = g >> 10;
    const int j = g & 1023;
    const float bz = bias[j];
    const float bf = bias[Hdim + j];
    const float bo = bias[2 * Hdim + j];
    float h = h0[g];
    for (int t = 0; t < Tdim; ++t) {
        const size_t base = (size_t)(b * Tdim + t) * N3H;
        const float z = tanhf(co[base + j] + bz);
        const float f = 1.0f / (1.0f + __expf(-(co[base + Hdim + j] + bf)));
        const float o = 1.0f / (1.0f + __expf(-(co[base + 2 * Hdim + j] + bo)));
        h = o * (f * h + (1.0f - f) * z);
        hbuf[(size_t)(b * Tdim + t) * Hdim + j] = h;
    }
    h0[g] = h;
}

// ---------------------------------------------------------------------------
__global__ __launch_bounds__(256) void ln_skip_kernel(const float* __restrict__ hbuf,
                                                      const float* __restrict__ gamma,
                                                      const float* __restrict__ beta,
                                                      const float* __restrict__ prevOut,
                                                      float* __restrict__ outBuf,
                                                      float* __restrict__ xBuf,
                                                      int hasSkip) {
    const int r = blockIdx.x;                       // 0 .. B*T-1
    const int tid = threadIdx.x;
    const float* row = hbuf + (size_t)r * Hdim;
    float vals[4], s = 0.0f, s2 = 0.0f;
    #pragma unroll
    for (int i = 0; i < 4; ++i) {
        const float v = row[tid + i * 256];
        vals[i] = v; s += v; s2 += v * v;
    }
    #pragma unroll
    for (int off = 16; off > 0; off >>= 1) {
        s  += __shfl_down(s, off);
        s2 += __shfl_down(s2, off);
    }
    __shared__ float rs[8], rs2[8];
    if ((tid & 31) == 0) { rs[tid >> 5] = s; rs2[tid >> 5] = s2; }
    __syncthreads();
    float ts = 0.0f, ts2 = 0.0f;
    #pragma unroll
    for (int i = 0; i < 8; ++i) { ts += rs[i]; ts2 += rs2[i]; }
    const float mu  = ts * (1.0f / 1024.0f);
    const float var = ts2 * (1.0f / 1024.0f) - mu * mu;
    const float inv = rsqrtf(var + 1e-5f);
    #pragma unroll
    for (int i = 0; i < 4; ++i) {
        const int c = tid + i * 256;
        const float o = (vals[i] - mu) * inv * gamma[c] + beta[c];
        outBuf[(size_t)r * Hdim + c] = o;
        xBuf[(size_t)r * Hdim + c] = o + (hasSkip ? prevOut[(size_t)r * Hdim + c] : 0.0f);
    }
}

// ---------------------------------------------------------------------------
__global__ __launch_bounds__(256) void lengths_kernel(const int* __restrict__ mask,
                                                      int* __restrict__ lengths) {
    const int b = blockIdx.x;
    const int tid = threadIdx.x;
    int s = 0;
    for (int t = tid; t < Tdim; t += 256) s += mask[b * Tdim + t];
    #pragma unroll
    for (int off = 16; off > 0; off >>= 1) s += __shfl_down(s, off);
    __shared__ int ps[8];
    if ((tid & 31) == 0) ps[tid >> 5] = s;
    __syncthreads();
    if (tid == 0) {
        int tot = 0;
        #pragma unroll
        for (int i = 0; i < 8; ++i) tot += ps[i];
        lengths[b] = tot;
    }
}

// ---------------------------------------------------------------------------
__global__ __launch_bounds__(256) void pool_kernel(const float* __restrict__ x,
                                                   const int* __restrict__ lengths,
                                                   float* __restrict__ pooled) {
    const int g = blockIdx.x * 256 + threadIdx.x;   // 0 .. B*H-1
    const int b = g >> 10;
    const int j = g & 1023;
    const int len = lengths[b];
    float m = -INFINITY;
    for (int t = 0; t < len; ++t)
        m = fmaxf(m, x[(size_t)(b * Tdim + t) * Hdim + j]);
    pooled[g] = m;
}

// ---------------------------------------------------------------------------
__global__ __launch_bounds__(32) void mlp1_wmma_kernel(const float* __restrict__ pooled,
                                                       const float* __restrict__ W1,
                                                       const float* __restrict__ b1,
                                                       float* __restrict__ h1) {
    const int lane = threadIdx.x;
    const int n0 = blockIdx.x * 64;
    const int hl = lane >> 4;
    const int m  = lane & 15;
    v8f acc[4] = {};
    for (int k = 0; k < Hdim; k += 4) {
        v2f a;
        a.x = pooled[m * Hdim + k + 2 * hl + 0];
        a.y = pooled[m * Hdim + k + 2 * hl + 1];
        #pragma unroll
        for (int nt = 0; nt < 4; ++nt) {
            v2f bf;
            bf.x = W1[(size_t)(k + 2 * hl + 0) * Hdim + n0 + nt * 16 + m];
            bf.y = W1[(size_t)(k + 2 * hl + 1) * Hdim + n0 + nt * 16 + m];
            acc[nt] = __builtin_amdgcn_wmma_f32_16x16x4_f32(
                false, a, false, bf, (short)0, acc[nt], false, false);
        }
    }
    #pragma unroll
    for (int nt = 0; nt < 4; ++nt) {
        const int col = n0 + nt * 16 + m;
        #pragma unroll
        for (int v = 0; v < 8; ++v) {
            const int row = v + 8 * hl;
            h1[row * Hdim + col] = fmaxf(acc[nt][v] + b1[col], 0.0f);
        }
    }
}

// ---------------------------------------------------------------------------
__global__ __launch_bounds__(32) void mlp2_kernel(const float* __restrict__ h1,
                                                  const float* __restrict__ W2,
                                                  const float* __restrict__ b2,
                                                  float* __restrict__ out) {
    const int g = threadIdx.x;                      // 0..31
    const int b = g >> 1;
    const int o = g & 1;
    float s = b2[o];
    for (int c = 0; c < Hdim; ++c) s += h1[b * Hdim + c] * W2[c * 2 + o];
    out[g] = s;
}

// ---------------------------------------------------------------------------
extern "C" void kernel_launch(void* const* d_in, const int* in_sizes, int n_in,
                              void* d_out, int out_size, void* d_ws, size_t ws_size,
                              hipStream_t stream) {
    const int*   ids   = (const int*)d_in[0];
    const int*   amask = (const int*)d_in[1];
    const float* emb   = (const float*)d_in[2];
    const float* convW = (const float*)d_in[3];
    const float* convb = (const float*)d_in[4];
    const float* gamma = (const float*)d_in[5];
    const float* beta  = (const float*)d_in[6];
    const float* W1    = (const float*)d_in[7];
    const float* b1    = (const float*)d_in[8];
    const float* W2    = (const float*)d_in[9];
    const float* b2    = (const float*)d_in[10];
    float* out = (float*)d_out;

    const size_t SZ = (size_t)Bdim * Tdim * Hdim;       // 16M elements
    const size_t WN = (size_t)NLay * 3 * Hdim * N3H;    // 28.3M weight elems
    const int    BH = Bdim * Hdim;

    float* x      = (float*)d_ws;
    float* hbuf   = x + SZ;
    float* outA   = hbuf + SZ;
    float* outB   = outA + SZ;
    float* co     = outB + SZ;                          // 3*SZ floats
    float* h0     = co + 3 * SZ;
    float* pooled = h0 + BH;
    float* h1     = pooled + BH;
    int*   lens   = (int*)(h1 + BH);
    f16*   xh     = (f16*)(lens + 64);
    f16*   xl     = xh + SZ;
    f16*   Wth    = xl + SZ;                            // transposed (ik,n,c)
    f16*   Wtl    = Wth + WN;

    embed_kernel<<<Bdim * Tdim, 256, 0, stream>>>(ids, emb, x);
    zero_kernel<<<BH / 256, 256, 0, stream>>>(h0, BH);
    lengths_kernel<<<Bdim, 256, 0, stream>>>(amask, lens);
    {
        dim3 tg(N3H / 32, Hdim / 32, NLay * 3);
        wsplit_transpose_kernel<<<tg, 256, 0, stream>>>(convW, Wth, Wtl);
    }

    for (int i = 0; i < NLay; ++i) {
        split_f16_kernel<<<(int)(SZ / 256), 256, 0, stream>>>(x, xh, xl, (int)SZ);
        dim3 grid(Bdim * Tdim / 64, N3H / 128);
        conv_wmma_f16_kernel<<<grid, 256, 0, stream>>>(
            xh, xl, Wth + (size_t)i * 3 * N3H * Hdim, Wtl + (size_t)i * 3 * N3H * Hdim, co);
        qrnn_scan_kernel<<<BH / 256, 256, 0, stream>>>(co, convb + i * N3H, h0, hbuf);
        const float* prev = (i == 0) ? nullptr : ((i == 1) ? outA : outB);
        float* cur = (i % 2 == 0) ? outA : outB;
        ln_skip_kernel<<<Bdim * Tdim, 256, 0, stream>>>(
            hbuf, gamma + i * Hdim, beta + i * Hdim, prev, cur, x, i > 0 ? 1 : 0);
    }

    pool_kernel<<<BH / 256, 256, 0, stream>>>(x, lens, pooled);
    mlp1_wmma_kernel<<<Hdim / 64, 32, 0, stream>>>(pooled, W1, b1, h1);
    mlp2_kernel<<<1, 32, 0, stream>>>(h1, W2, b2, out);
}
